// SweepGaterV3_83571473645671
// MI455X (gfx1250) — matrix operand order repaired and checked by reading, not compiled, not run on hardware
//
#include <hip/hip_runtime.h>
#include <hip/hip_bf16.h>
#include <stdint.h>

typedef __bf16 bf16;
typedef __attribute__((ext_vector_type(16))) __bf16 v16bf;
typedef __attribute__((ext_vector_type(8)))  float  v8f;

// ---- problem constants (fixed by reference) ----
constexpr int Bb  = 2;
constexpr int Sn  = 5;
constexpr int C   = 64;
constexpr int Hh  = 192;
constexpr int Ww  = 192;
constexpr int HW  = Hh * Ww;        // 36864
constexpr int Np  = Bb * HW;        // 73728 pixels
constexpr int NMAT = 11;            // MS, MT[5], D[5]
constexpr int NTILES = Np / 16;     // 4608 16-pixel WMMA tiles

// ---- workspace layout (bytes) ----
constexpr size_t WS_ACC     = 0;                      // 16 f32 (sum delta^2 per s)
constexpr size_t WS_SCORE   = 64;                     // 16 f32 (heuristic score per s)
constexpr size_t WS_CVEC    = 128;                    // 64 f32 folded bias
constexpr size_t WS_FRAG    = 512;                    // bf16 A-fragments, 11 * 4096 elems
constexpr size_t FRAG_BYTES = (size_t)NMAT * 4096 * 2; // 90112
constexpr size_t WS_W2FRAG  = WS_FRAG + FRAG_BYTES;   // 1024 bf16 = 2048 B
constexpr size_t WS_SL      = WS_W2FRAG + 2048;       // 5 * Np f32 score_learn

__device__ __forceinline__ v8f zero8() {
  v8f z;
#pragma unroll
  for (int r = 0; r < 8; ++r) z[r] = 0.f;
  return z;
}

// =====================================================================
// Kernel 1: fold constants into bf16 WMMA A-fragments + cvec, zero acc.
//   MS   = sum_s (W1S_s - W1D_s) * D_s
//   MT_s = (W1T_s + W1D_s) * D_s        D_s = diag(1/sigma_s) * A_s
//   cvec = b1 + sum_s (W1S_s + W1T_s) * e_s,  e_s = (b_s - mu_s)/sigma_s
// A-fragment layout (16-bit A 16x32): lane = hi*16 + (M&15);
//   elem j : K = kb*32 + hi*16 + j
// =====================================================================
__global__ void fold_kernel(const float* __restrict__ adapt_w,
                            const float* __restrict__ adapt_b,
                            const float* __restrict__ r_w1,
                            const float* __restrict__ r_b1,
                            const float* __restrict__ r_w2,
                            const float* __restrict__ ema_mu,
                            const float* __restrict__ ema_sigma,
                            char* __restrict__ ws) {
  const int t = threadIdx.x;
  float* acc  = (float*)(ws + WS_ACC);
  float* cvec = (float*)(ws + WS_CVEC);
  bf16*  frag = (bf16*)(ws + WS_FRAG);
  bf16*  w2f  = (bf16*)(ws + WS_W2FRAG);

  if (t < 16) acc[t] = 0.f;

  if (t < C) {  // folded bias vector
    float cv = r_b1[t];
    for (int s = 0; s < Sn; ++s)
      for (int d = 0; d < C; ++d) {
        float e = (adapt_b[s * C + d] - ema_mu[s * C + d]) / ema_sigma[s * C + d];
        cv += (r_w1[t * 960 + s * 192 + d] + r_w1[t * 960 + s * 192 + 64 + d]) * e;
      }
    cvec[t] = cv;
  }

  for (int idx = t; idx < NMAT * 4096; idx += blockDim.x) {
    const int j    = idx & 15;
    const int lane = (idx >> 4) & 31;
    const int kb   = (idx >> 9) & 1;
    const int rb   = (idx >> 10) & 3;
    const int mat  = idx >> 12;
    const int row  = rb * 16 + (lane & 15);
    const int hi   = lane >> 4;
    const int k    = kb * 32 + hi * 16 + j;
    float v;
    if (mat == 0) {                       // MS
      v = 0.f;
      for (int s = 0; s < Sn; ++s)
        for (int d = 0; d < C; ++d)
          v += (r_w1[row * 960 + s * 192 + d] - r_w1[row * 960 + s * 192 + 128 + d]) *
               adapt_w[s * 4096 + d * 64 + k] / ema_sigma[s * C + d];
    } else if (mat <= 5) {                // MT_s
      const int s = mat - 1;
      v = 0.f;
      for (int d = 0; d < C; ++d)
        v += (r_w1[row * 960 + s * 192 + 64 + d] + r_w1[row * 960 + s * 192 + 128 + d]) *
             adapt_w[s * 4096 + d * 64 + k] / ema_sigma[s * C + d];
    } else {                              // D_s
      const int s = mat - 6;
      v = adapt_w[s * 4096 + row * 64 + k] / ema_sigma[s * C + row];
    }
    frag[idx] = (bf16)v;
  }

  // r_w2 as 16x64 A operand (rows >= 5 zero-padded)
  for (int idx = t; idx < 1024; idx += blockDim.x) {
    const int j    = idx & 15;
    const int lane = (idx >> 4) & 31;
    const int kb   = idx >> 9;
    const int row  = lane & 15;
    const int hi   = lane >> 4;
    const int k    = kb * 32 + hi * 16 + j;
    w2f[idx] = (bf16)((row < Sn) ? r_w2[row * 64 + k] : 0.f);
  }
}

// =====================================================================
// Kernel 2: heavy pass. One wave = exactly one 16-pixel tile
// (576 blocks x 8 waves = 4608 tiles).
//   h      = relu(MS*S + sum_s MT_s*T_s + cvec)      (bf16 WMMA, K=64)
//   ssq_s += |D_s*(T_s - S)|^2 per pixel             (bf16 WMMA)
//   score_learn = W2 * h                             (bf16 WMMA)
// Fragment bank (88 KB) pulled into LDS with async-to-LDS DMA.
// The s-loop is kept ROLLED to bound VGPR pressure (<256, no spills).
// =====================================================================
__global__ void __launch_bounds__(256) gate_main_kernel(
    const float* __restrict__ S, const float* __restrict__ T,
    char* __restrict__ ws) {
  extern __shared__ v16bf smemv[];   // NMAT*4*2*32 fragments of 32B
  const int tid    = threadIdx.x;
  const int lane   = tid & 31;
  const int waveId = tid >> 5;
  const int nIdx   = lane & 15;
  const int hi     = lane >> 4;

  // ---- async fill of LDS fragment bank (CDNA5 GLOBAL_LOAD_ASYNC_TO_LDS) ----
  {
    const uint64_t sbase   = (uint64_t)(uintptr_t)(ws + WS_FRAG);
    const uint32_t ldsBase = (uint32_t)(uintptr_t)smemv;  // low 32 bits = LDS offset
    for (int i = tid; i < (int)(FRAG_BYTES / 16); i += 256) {
      const uint32_t voff   = (uint32_t)i * 16u;
      const uint32_t ldsoff = ldsBase + voff;
      asm volatile("global_load_async_to_lds_b128 %0, %1, %2"
                   :: "v"(ldsoff), "v"(voff), "s"(sbase) : "memory");
    }
    asm volatile("s_wait_asynccnt 0" ::: "memory");
    __syncthreads();
  }

  float* acc = (float*)(ws + WS_ACC);
  float* sl  = (float*)(ws + WS_SL);

  const int tile = blockIdx.x * 8 + waveId;       // one tile per wave
  const int gp0  = tile * 16;
  const int b    = gp0 / HW;
  const int p    = (gp0 % HW) + nIdx;             // this lane's pixel column

  // ---- load S column, build B fragments (32x16, K=channels, N=pixels) ----
  float SF[32];
  v16bf SB[2];
#pragma unroll
  for (int kb = 0; kb < 2; ++kb)
#pragma unroll
    for (int j = 0; j < 16; ++j) {
      const int c = kb * 32 + hi * 16 + j;
      const float v = S[(size_t)(b * C + c) * HW + p];
      SF[kb * 16 + j] = v;
      SB[kb][j] = (bf16)v;
    }

  v8f hac[4] = {zero8(), zero8(), zero8(), zero8()};

  // h += MS * S
#pragma unroll
  for (int rb = 0; rb < 4; ++rb) {
    const v16bf a0 = smemv[((0 * 4 + rb) * 2 + 0) * 32 + lane];
    const v16bf a1 = smemv[((0 * 4 + rb) * 2 + 1) * 32 + lane];
    hac[rb] = __builtin_amdgcn_wmma_f32_16x16x32_bf16(false, a0, false, SB[0], (short)0, hac[rb], false, false);
    hac[rb] = __builtin_amdgcn_wmma_f32_16x16x32_bf16(false, a1, false, SB[1], (short)0, hac[rb], false, false);
  }

  float ssq[Sn] = {0.f, 0.f, 0.f, 0.f, 0.f};

  // Rolled source loop: bounds VGPR live set, avoids spills / vgpr-msb mode.
#pragma unroll 1
  for (int s = 0; s < Sn; ++s) {
    const float* Tsrc = T + (size_t)((b * Sn + s) * C) * HW + p;
    v16bf TB[2], DB[2];
#pragma unroll
    for (int kb = 0; kb < 2; ++kb)
#pragma unroll
      for (int j = 0; j < 16; ++j) {
        const int c = kb * 32 + hi * 16 + j;
        const float tv = Tsrc[(size_t)c * HW];
        TB[kb][j] = (bf16)tv;
        DB[kb][j] = (bf16)(tv - SF[kb * 16 + j]);
      }
    float s2 = 0.f;
#pragma unroll
    for (int rb = 0; rb < 4; ++rb) {
      const v16bf mt0 = smemv[(((1 + s) * 4 + rb) * 2 + 0) * 32 + lane];
      const v16bf mt1 = smemv[(((1 + s) * 4 + rb) * 2 + 1) * 32 + lane];
      hac[rb] = __builtin_amdgcn_wmma_f32_16x16x32_bf16(false, mt0, false, TB[0], (short)0, hac[rb], false, false);
      hac[rb] = __builtin_amdgcn_wmma_f32_16x16x32_bf16(false, mt1, false, TB[1], (short)0, hac[rb], false, false);

      const v16bf d0 = smemv[(((6 + s) * 4 + rb) * 2 + 0) * 32 + lane];
      const v16bf d1 = smemv[(((6 + s) * 4 + rb) * 2 + 1) * 32 + lane];
      v8f dz = zero8();
      dz = __builtin_amdgcn_wmma_f32_16x16x32_bf16(false, d0, false, DB[0], (short)0, dz, false, false);
      dz = __builtin_amdgcn_wmma_f32_16x16x32_bf16(false, d1, false, DB[1], (short)0, dz, false, false);
#pragma unroll
      for (int r = 0; r < 8; ++r) s2 += dz[r] * dz[r];
    }
    ssq[s] = s2;
  }

  // ---- bias + ReLU (cvec loaded at point of use; L2-resident, tiny) ----
  const float* cvec = (const float*)(ws + WS_CVEC);
#pragma unroll
  for (int rb = 0; rb < 4; ++rb)
#pragma unroll
    for (int r = 0; r < 8; ++r)
      hac[rb][r] = fmaxf(hac[rb][r] + cvec[rb * 16 + r + 8 * hi], 0.f);

  // ---- score_learn = W2 * h via WMMA: D-layout -> B-layout (shfl_xor 16) ----
  v16bf HB0, HB1;
#pragma unroll
  for (int r = 0; r < 8; ++r) {
    const float a0v = hac[0][r], a1v = hac[1][r], a2v = hac[2][r], a3v = hac[3][r];
    const float p0 = __shfl_xor(a0v, 16, 32);
    const float p1 = __shfl_xor(a1v, 16, 32);
    const float p2 = __shfl_xor(a2v, 16, 32);
    const float p3 = __shfl_xor(a3v, 16, 32);
    HB0[r]     = (bf16)(hi ? p1 : a0v);
    HB0[8 + r] = (bf16)(hi ? a1v : p0);
    HB1[r]     = (bf16)(hi ? p3 : a2v);
    HB1[8 + r] = (bf16)(hi ? a3v : p2);
  }
  const v16bf* w2fG = (const v16bf*)(ws + WS_W2FRAG);
  const v16bf  w2A0 = w2fG[0 * 32 + lane];
  const v16bf  w2A1 = w2fG[1 * 32 + lane];
  v8f sac = zero8();
  sac = __builtin_amdgcn_wmma_f32_16x16x32_bf16(false, w2A0, false, HB0, (short)0, sac, false, false);
  sac = __builtin_amdgcn_wmma_f32_16x16x32_bf16(false, w2A1, false, HB1, (short)0, sac, false, false);

  if (hi == 0) {   // rows j=0..4 live in lanes 0-15, regs 0..4
#pragma unroll
    for (int j = 0; j < Sn; ++j)
      sl[(size_t)j * Np + gp0 + nIdx] = sac[j];
  }

  // ---- reduce per-source sum(delta^2) across the wave, accumulate ----
#pragma unroll
  for (int s = 0; s < Sn; ++s) {
    float v = ssq[s];
    for (int m = 16; m >= 1; m >>= 1) v += __shfl_xor(v, m, 32);
    if (lane == 0) atomicAdd(&acc[s], v);
  }
}

// =====================================================================
// Kernel 3: finalize heuristic score (advantage + band penalty)
// =====================================================================
__global__ void score_kernel(const float* __restrict__ ema_proxy,
                             char* __restrict__ ws) {
  if (threadIdx.x != 0 || blockIdx.x != 0) return;
  const float* acc = (const float*)(ws + WS_ACC);
  float* score = (float*)(ws + WS_SCORE);
  float cur[Sn], impr[Sn], mean = 0.f;
  const float inv = 1.f / ((float)C * (float)Np);
  for (int s = 0; s < Sn; ++s) {
    cur[s]  = acc[s] * inv;
    impr[s] = ema_proxy[s] - cur[s];
    mean   += impr[s];
  }
  mean *= (1.f / Sn);
  for (int s = 0; s < Sn; ++s) {
    const float below = fmaxf(0.05f - cur[s], 0.f);
    const float above = fmaxf(cur[s] - 0.2f, 0.f);
    const float band  = -(below * below + above * above);
    score[s] = 1.0f * (impr[s] - mean) + 0.5f * band;   // ALPHA_ADV, BETA_BAND
  }
}

// =====================================================================
// Kernel 4: per-pixel 5-way softmax at TEMP=0.7
// =====================================================================
__global__ void softmax_kernel(const float* __restrict__ r_b2,
                               const char* __restrict__ ws,
                               float* __restrict__ out) {
  const int gp = blockIdx.x * blockDim.x + threadIdx.x;
  if (gp >= Np) return;
  const float* score = (const float*)(ws + WS_SCORE);
  const float* sl    = (const float*)(ws + WS_SL);
  const int b = gp / HW, p = gp % HW;
  float lg[Sn], m = -1e30f;
#pragma unroll
  for (int s = 0; s < Sn; ++s) {
    lg[s] = (0.5f * score[s] + 0.5f * (sl[(size_t)s * Np + gp] + r_b2[s])) * (1.f / 0.7f);
    m = fmaxf(m, lg[s]);
  }
  float sum = 0.f, e[Sn];
#pragma unroll
  for (int s = 0; s < Sn; ++s) { e[s] = __expf(lg[s] - m); sum += e[s]; }
  const float rs = 1.f / sum;
#pragma unroll
  for (int s = 0; s < Sn; ++s)
    out[(size_t)(b * Sn + s) * HW + p] = e[s] * rs;
}

// =====================================================================
extern "C" void kernel_launch(void* const* d_in, const int* in_sizes, int n_in,
                              void* d_out, int out_size, void* d_ws, size_t ws_size,
                              hipStream_t stream) {
  const float* S          = (const float*)d_in[0];
  const float* T          = (const float*)d_in[1];
  const float* adapt_w    = (const float*)d_in[2];
  const float* adapt_b    = (const float*)d_in[3];
  const float* r_w1       = (const float*)d_in[4];
  const float* r_b1       = (const float*)d_in[5];
  const float* r_w2       = (const float*)d_in[6];
  const float* r_b2       = (const float*)d_in[7];
  const float* ema_mu     = (const float*)d_in[8];
  const float* ema_sigma  = (const float*)d_in[9];
  const float* ema_proxy  = (const float*)d_in[10];
  char*  ws  = (char*)d_ws;
  float* out = (float*)d_out;

  hipLaunchKernelGGL(fold_kernel, dim3(1), dim3(256), 0, stream,
                     adapt_w, adapt_b, r_w1, r_b1, r_w2, ema_mu, ema_sigma, ws);
  hipLaunchKernelGGL(gate_main_kernel, dim3(NTILES / 8), dim3(256), FRAG_BYTES, stream,
                     S, T, ws);
  hipLaunchKernelGGL(score_kernel, dim3(1), dim3(32), 0, stream, ema_proxy, ws);
  hipLaunchKernelGGL(softmax_kernel, dim3((Np + 255) / 256), dim3(256), 0, stream,
                     r_b2, ws, out);
}
